// AttnBlock_69466801045525
// MI455X (gfx1250) — compile-verified, matrix operand order checked by reading, not run on hardware
//
#include <hip/hip_runtime.h>

// Problem constants: B=4, C=64, H=W=40, k=11 -> Hp=Wp=30, L=P=900, D=7744.
#define IMG 40
#define CHN 64
#define KW  11
#define HP  30
#define LP  900

typedef __attribute__((ext_vector_type(16))) _Float16 v16h;
typedef __attribute__((ext_vector_type(8)))  _Float16 v8h;
typedef __attribute__((ext_vector_type(8)))  float    v8f;

// Get the 32-bit LDS byte offset of a __shared__ pointer (addrspace(3)
// ptrtoint; folds to the raw offset at -O3). Needed for async-to-LDS ops,
// whose VDST VGPR holds an offset relative to the wave's LDS base.
__device__ __forceinline__ unsigned lds_off(const void* p) {
  return (unsigned)(unsigned long long)
         (__attribute__((address_space(3))) const char*)p;
}

// One per-lane async 16B copy: global memory -> LDS, tracked by ASYNCcnt.
__device__ __forceinline__ void async_copy_b128(unsigned ldsByteOff,
                                                const void* gptr) {
  unsigned long long ga = (unsigned long long)gptr;
  asm volatile("global_load_async_to_lds_b128 %0, %1, off"
               :: "v"(ldsByteOff), "v"(ga) : "memory");
}

__device__ __forceinline__ void wait_async0() {
  asm volatile("s_wait_asynccnt 0" ::: "memory");
}

// ---------------------------------------------------------------------------
// Kernel 0: [B,C,H,W] f32 -> [B,H,W,C] f16 (channels-last). One-time, tiny.
// ---------------------------------------------------------------------------
__global__ void transpose_cvt(const float* __restrict__ im1,
                              const float* __restrict__ im2,
                              _Float16* __restrict__ im1t,
                              _Float16* __restrict__ im2t) {
  int idx = blockIdx.x * 256 + threadIdx.x;      // enumerates ((b*40+y)*40+x)*64+c
  if (idx >= 4 * IMG * IMG * CHN) return;
  int c    = idx & 63;
  int rest = idx >> 6;                           // b*1600 + y*40 + x
  int x = rest % IMG;
  int t = rest / IMG;
  int y = t % IMG;
  int b = t / IMG;
  size_t src = ((size_t)(b * CHN + c) * IMG + y) * IMG + x;
  im1t[idx] = (_Float16)im1[src];
  im2t[idx] = (_Float16)im2[src];
}

// ---------------------------------------------------------------------------
// Kernel 1: per-patch inverse L2 norms of im1 patches. One wave per patch.
// ---------------------------------------------------------------------------
__global__ void norms_kernel(const _Float16* __restrict__ im1t,
                             float* __restrict__ invn) {
  int tid  = threadIdx.x;
  int wid  = tid >> 5;
  int lane = tid & 31;
  int patch = blockIdx.x * 4 + wid;              // [0, 3600)
  int b = patch / LP, l = patch - b * LP;
  int ly = l / HP, lx = l - ly * HP;
  int ssub = lane >> 3, chk = lane & 7;          // 4 offsets x 8 chunks / pass
  float sum = 0.f;
  for (int s0 = 0; s0 < 121; s0 += 4) {
    int s = s0 + ssub;
    if (s < 121) {
      int dy = s / KW, dx = s - dy * KW;
      const v8h vv = *(const v8h*)(im1t +
          (((size_t)b * IMG + ly + dy) * IMG + (lx + dx)) * CHN + chk * 8);
#pragma unroll
      for (int i = 0; i < 8; i++) { float f = (float)vv[i]; sum += f * f; }
    }
  }
#pragma unroll
  for (int off = 1; off < 32; off <<= 1) sum += __shfl_xor(sum, off, 32);
  if (lane == 0) invn[patch] = 1.0f / fmaxf(sqrtf(sum), 1e-4f);
}

// ---------------------------------------------------------------------------
// Kernel 2: initialize output for atomic-max epilogue.
// ---------------------------------------------------------------------------
__global__ void init_out(float* __restrict__ out) {
  int i = blockIdx.x * 256 + threadIdx.x;
  if (i < 4 * LP) out[i] = -3.402823466e38f;     // -FLT_MAX
}

// Sign-split float atomic max (works with -FLT_MAX init; no NaNs here).
__device__ inline void atomicMaxFloat(float* addr, float val) {
  if (val >= 0.f) atomicMax((int*)addr, __float_as_int(val));
  else            atomicMin((unsigned int*)addr, __float_as_uint(val));
}

// Build one 16x32 f16 WMMA operand fragment from an LDS tile stored as
// 64 rows x 64 halves (128B rows), 16B chunks XOR-swizzled by (row&7).
// Per ISA: lanes 0-15 hold K{0..7,16..23}, lanes 16-31 hold K{8..15,24..31}
// => two ds_load_b128 at chunk (j*4+hi) and (j*4+hi+2).
__device__ __forceinline__ v16h ldfrag(const _Float16* sm, int rowBase,
                                       int j, int lane) {
  int r  = rowBase + (lane & 15);
  int hi = lane >> 4;
  int c0 = j * 4 + hi;
  int sw = r & 7;
  v8h lo = *(const v8h*)(sm + r * 64 + (((c0    ) ^ sw) << 3));
  v8h hh = *(const v8h*)(sm + r * 64 + (((c0 + 2) ^ sw) << 3));
  v16h o;
#pragma unroll
  for (int i = 0; i < 8; i++) { o[i] = lo[i]; o[i + 8] = hh[i]; }
  return o;
}

// ---------------------------------------------------------------------------
// Kernel 3: WMMA correlation GEMM + scale + row-max.
// Grid (15 l-tiles, 15 p-tiles, 4 batches), 128 threads = 4 waves.
// 64x64 output tile; each wave: 32x32 sub-tile = 2x2 f32 accumulators.
// Double-buffered LDS; tile s+1 streamed via GLOBAL_LOAD_ASYNC_TO_LDS_B128
// (ASYNCcnt) while tile s feeds 8 v_wmma; one barrier per K-iteration.
// ---------------------------------------------------------------------------
__global__ __launch_bounds__(128) void corr_gemm(
    const _Float16* __restrict__ im1t,   // filter side (l columns)
    const _Float16* __restrict__ im2t,   // data side   (p rows)
    const float* __restrict__ invn,
    float* __restrict__ out) {
  __shared__ __align__(16) _Float16 Asm[2][64 * 64];  // p-tile, 2 x 8KB
  __shared__ __align__(16) _Float16 Bsm[2][64 * 64];  // l-tile, 2 x 8KB
  __shared__ float sInv[64];

  const int tid   = threadIdx.x;
  const int lane  = tid & 31;
  const int wid   = tid >> 5;
  const int waveP = wid >> 1;            // 0..1 : p strip
  const int waveL = wid & 1;             // 0..1 : l strip
  const int b     = blockIdx.z;
  const int pBase = blockIdx.y * 64;
  const int lBase = blockIdx.x * 64;

  // Staging coordinates: 512 16B chunks per tile, 128 threads -> 4 slots each.
  size_t gA[4], gB[4];
  unsigned aL[4], bL[4];
#pragma unroll
  for (int k = 0; k < 4; k++) {
    int idx = tid + k * 128;
    int r = idx >> 3, ch = idx & 7;
    int pp = pBase + r; if (pp > LP - 1) pp = LP - 1;   // clamp (masked later)
    int py = pp / HP, px = pp - py * HP;
    gA[k] = (((size_t)b * IMG + py) * IMG + px) * CHN + ch * 8;
    int ll = lBase + r; if (ll > LP - 1) ll = LP - 1;
    int ly = ll / HP, lx = ll - ly * HP;
    gB[k] = (((size_t)b * IMG + ly) * IMG + lx) * CHN + ch * 8;
    int ld = r * 64 + ((ch ^ (r & 7)) << 3);            // swizzled LDS offset
    aL[k] = lds_off(&Asm[0][ld]);
    bL[k] = lds_off(&Bsm[0][ld]);
  }

  // Prologue: stream tile s=0 into buffer 0 (offset 0: dy=dx=0).
#pragma unroll
  for (int k = 0; k < 4; k++) {
    async_copy_b128(aL[k], im2t + gA[k]);
    async_copy_b128(bL[k], im1t + gB[k]);
  }
  if (tid < 64) {
    int l = lBase + tid; if (l > LP - 1) l = LP - 1;
    sInv[tid] = invn[b * LP + l];
  }
  wait_async0();
  __syncthreads();

  v8f acc00 = {}, acc01 = {}, acc10 = {}, acc11 = {};

  for (int s = 0; s < 121; s++) {
    // Prefetch tile s+1 into the other buffer while buffer s&1 feeds WMMA.
    if (s < 120) {
      int s1 = s + 1;
      int dy = s1 / KW, dx = s1 - dy * KW;
      size_t off = ((size_t)dy * IMG + dx) * CHN;
      unsigned bsel = (unsigned)(s1 & 1) * (64 * 64 * 2);  // buffer byte stride
#pragma unroll
      for (int k = 0; k < 4; k++) {
        async_copy_b128(aL[k] + bsel, im2t + gA[k] + off);
        async_copy_b128(bL[k] + bsel, im1t + gB[k] + off);
      }
    }
    const _Float16* As = Asm[s & 1];
    const _Float16* Bs = Bsm[s & 1];
#pragma unroll
    for (int j = 0; j < 2; j++) {                      // K = 64 -> 2 x K32
      v16h a0 = ldfrag(As, waveP * 32,      j, lane);
      v16h a1 = ldfrag(As, waveP * 32 + 16, j, lane);
      v16h b0 = ldfrag(Bs, waveL * 32,      j, lane);
      v16h b1 = ldfrag(Bs, waveL * 32 + 16, j, lane);
      acc00 = __builtin_amdgcn_wmma_f32_16x16x32_f16(false, a0, false, b0, (short)0, acc00, false, false);
      acc01 = __builtin_amdgcn_wmma_f32_16x16x32_f16(false, a0, false, b1, (short)0, acc01, false, false);
      acc10 = __builtin_amdgcn_wmma_f32_16x16x32_f16(false, a1, false, b0, (short)0, acc10, false, false);
      acc11 = __builtin_amdgcn_wmma_f32_16x16x32_f16(false, a1, false, b1, (short)0, acc11, false, false);
    }
    // Drain our async writes to buf[s+1&1], then one barrier: all waves have
    // finished reading buf[s&1] and all prefetched data is visible.
    wait_async0();
    __syncthreads();
  }

  // Epilogue: C layout => lane column l fixed per 16x16 tile; scale by invn,
  // mask padded l with -BIG, shfl-max over the 16-lane half, atomic-max out.
  const int col16 = lane & 15;
  const int hi    = lane >> 4;
  const float NEG = -3.0e38f;
  float inv0 = sInv[waveL * 32 + col16];
  float inv1 = sInv[waveL * 32 + 16 + col16];
  bool ok0 = (lBase + waveL * 32 + col16)      < LP;
  bool ok1 = (lBase + waveL * 32 + 16 + col16) < LP;
#pragma unroll
  for (int ai = 0; ai < 2; ai++) {
    v8f cA = ai ? acc10 : acc00;   // li = 0
    v8f cB = ai ? acc11 : acc01;   // li = 1
#pragma unroll
    for (int v = 0; v < 8; v++) {
      float m  = ok0 ? cA[v] * inv0 : NEG;
      float m2 = ok1 ? cB[v] * inv1 : NEG;
      m = fmaxf(m, m2);
#pragma unroll
      for (int o = 1; o < 16; o <<= 1) m = fmaxf(m, __shfl_xor(m, o, 32));
      if (col16 == 0) {
        int p = pBase + waveP * 32 + ai * 16 + v + hi * 8;
        if (p < LP) atomicMaxFloat(out + b * LP + p, m);
      }
    }
  }
}

// ---------------------------------------------------------------------------
// Launch. Workspace layout (needs ~1.66 MB):
//   im1t: 409600 halves | im2t: 409600 halves | invn: 3600 floats
// ---------------------------------------------------------------------------
extern "C" void kernel_launch(void* const* d_in, const int* in_sizes, int n_in,
                              void* d_out, int out_size, void* d_ws, size_t ws_size,
                              hipStream_t stream) {
  (void)in_sizes; (void)n_in; (void)out_size; (void)ws_size;
  const float* im1 = (const float*)d_in[0];
  const float* im2 = (const float*)d_in[1];
  float* out = (float*)d_out;

  _Float16* im1t = (_Float16*)d_ws;
  _Float16* im2t = im1t + 4 * IMG * IMG * CHN;
  float*    invn = (float*)(im2t + 4 * IMG * IMG * CHN);

  transpose_cvt<<<1600, 256, 0, stream>>>(im1, im2, im1t, im2t);
  norms_kernel<<<900, 128, 0, stream>>>(im1t, invn);
  init_out<<<15, 256, 0, stream>>>(out);
  dim3 grid(15, 15, 4);   // (l tiles, p tiles, batch)
  corr_gemm<<<grid, 128, 0, stream>>>(im1t, im2t, invn, out);
}
